// ScaledDotProductAttention_17222818857528
// MI455X (gfx1250) — matrix operand order repaired and checked by reading, not compiled, not run on hardware
//
#include <hip/hip_runtime.h>

// ---- CDNA5 WMMA fragment types -------------------------------------------
typedef _Float16 v16h __attribute__((ext_vector_type(16)));
typedef _Float16 h8  __attribute__((ext_vector_type(8)));
typedef _Float16 h4  __attribute__((ext_vector_type(4)));
typedef float    v8f __attribute__((ext_vector_type(8)));

union fragA { v16h v; h8 h[2]; };

// ---- problem constants ----------------------------------------------------
#define DHEAD   128
#define LQ      2048
#define SK      2048
#define NBH     64          // B*H
#define NTHR    256         // 8 waves per workgroup
#define NWAVES  8
#define BLOCK_M 128         // q rows per workgroup (16 per wave, 8 waves)
#define BLOCK_N 64          // k/v rows per S-tile
#define NTILES  (SK / BLOCK_N)
#define KPAD    136         // 128 + 8 halves   (row stride of K tile in LDS)
#define VPAD    72          // 64 + 8 halves    (row stride of V^T tile in LDS)
#define PPAD    72          // 64 + 8 halves    (row stride of P tile in LDS)

__device__ __forceinline__ h8 cvt8(const float* p) {
    float4 a = *(const float4*)(p);
    float4 b = *(const float4*)(p + 4);
    h8 r;
    r[0] = (_Float16)a.x; r[1] = (_Float16)a.y; r[2] = (_Float16)a.z; r[3] = (_Float16)a.w;
    r[4] = (_Float16)b.x; r[5] = (_Float16)b.y; r[6] = (_Float16)b.z; r[7] = (_Float16)b.w;
    return r;
}

__global__ __launch_bounds__(NTHR)
void fa_fwd_f16wmma(const float* __restrict__ Q, const float* __restrict__ K,
                    const float* __restrict__ V, const float* __restrict__ Mask,
                    float* __restrict__ O)
{
    // double-buffered K / V^T tiles; wave-private P and Q-fragment buffers
    __shared__ _Float16 Ksh [2][BLOCK_N * KPAD];     // K tile, row-major [s][d]
    __shared__ _Float16 Vtsh[2][DHEAD   * VPAD];     // V tile, transposed [d][s]
    __shared__ _Float16 Psh [NWAVES * 16 * PPAD];    // probs, per-wave [m][s]
    __shared__ _Float16 Qsh [NWAVES * 4 * 32 * 16];  // Q A-fragments, per-lane

    const int tid  = threadIdx.x;
    const int wave = tid >> 5;
    const int lane = tid & 31;
    const int ln   = lane & 15;     // N (or M) index within fragment
    const int hi   = lane >> 4;     // lane-half selector

    const int bh    = blockIdx.y;
    const int qrow0 = blockIdx.x * BLOCK_M + wave * 16;

    const float scale = 0.08838834764831845f;   // 1/sqrt(128)
    const float L2E   = 1.4426950408889634f;

    // ---- Q fragments (A-operand layout) -> wave-private LDS ---------------
    {
        const float* qrow = Q + ((size_t)bh * LQ + (qrow0 + ln)) * DHEAD;
        #pragma unroll
        for (int kc = 0; kc < 4; ++kc) {
            const int k0 = kc * 32 + hi * 8;
            h8 lo = cvt8(qrow + k0);
            h8 hi8 = cvt8(qrow + k0 + 16);
            _Float16* qs = &Qsh[(((wave << 2) + kc) * 32 + lane) * 16];
            *(h8*)(qs)     = lo;
            *(h8*)(qs + 8) = hi8;
        }
    }

    // ---- flash-attention state -------------------------------------------
    v8f acc[8];
    #pragma unroll
    for (int i = 0; i < 8; ++i)
        #pragma unroll
        for (int j = 0; j < 8; ++j) acc[i][j] = 0.0f;
    float mrow[8], lrow[8];
    #pragma unroll
    for (int v = 0; v < 8; ++v) { mrow[v] = -3.0e38f; lrow[v] = 0.0f; }

    const float* kgb = K + (size_t)bh * SK * DHEAD;
    const float* vgb = V + (size_t)bh * SK * DHEAD;

    // ---- prologue: stage tile 0 into buffer 0 ----------------------------
    {
        const float4* kg4 = (const float4*)kgb;
        const float4* vg4 = (const float4*)vgb;
        #pragma unroll
        for (int i = 0; i < 8; ++i) {
            const int fi  = tid + i * NTHR;
            const int row = fi >> 5;
            const int col = (fi & 31) << 2;
            float4 kk = kg4[fi];
            float4 vv = vg4[fi];
            h4 kh;
            kh[0] = (_Float16)kk.x; kh[1] = (_Float16)kk.y;
            kh[2] = (_Float16)kk.z; kh[3] = (_Float16)kk.w;
            *(h4*)(&Ksh[0][row * KPAD + col]) = kh;
            Vtsh[0][(col + 0) * VPAD + row] = (_Float16)vv.x;
            Vtsh[0][(col + 1) * VPAD + row] = (_Float16)vv.y;
            Vtsh[0][(col + 2) * VPAD + row] = (_Float16)vv.z;
            Vtsh[0][(col + 3) * VPAD + row] = (_Float16)vv.w;
        }
    }
    __syncthreads();

    for (int it = 0; it < NTILES; ++it) {
        const int s0  = it * BLOCK_N;
        const int cur = it & 1;
        const bool haveNext = (it + 1 < NTILES);

        const _Float16* Kc = Ksh[cur];
        const _Float16* Vc = Vtsh[cur];
        _Float16* Kn = Ksh[cur ^ 1];
        _Float16* Vn = Vtsh[cur ^ 1];

        // ---- phase 1: issue next K-tile loads (32 staging VGPRs) ---------
        float4 kreg[8];
        if (haveNext) {
            const float4* kg4 = (const float4*)(kgb + (size_t)(s0 + BLOCK_N) * DHEAD);
            #pragma unroll
            for (int i = 0; i < 8; ++i) kreg[i] = kg4[tid + i * NTHR];
            if (it + 2 < NTILES) {   // warm L2 for tile it+2
                const float* nk = kgb + (size_t)(s0 + 2 * BLOCK_N) * DHEAD;
                const float* nv = vgb + (size_t)(s0 + 2 * BLOCK_N) * DHEAD;
                __builtin_prefetch(nk + (size_t)tid * 32, 0, 1);
                __builtin_prefetch(nv + (size_t)tid * 32, 0, 1);
            }
        }

        // ---- S = Q * K^T  (16 x 64), f32 accum; kc-outer, nb-inner -------
        v8f sc[4];
        #pragma unroll
        for (int nb = 0; nb < 4; ++nb)
            #pragma unroll
            for (int j = 0; j < 8; ++j) sc[nb][j] = 0.0f;
        #pragma unroll
        for (int kc = 0; kc < 4; ++kc) {
            fragA a;
            const _Float16* qs = &Qsh[(((wave << 2) + kc) * 32 + lane) * 16];
            a.h[0] = *(const h8*)(qs);
            a.h[1] = *(const h8*)(qs + 8);
            #pragma unroll
            for (int nb = 0; nb < 4; ++nb) {
                fragA b;
                const int off = (nb * 16 + ln) * KPAD + kc * 32 + hi * 8;
                b.h[0] = *(const h8*)(&Kc[off]);
                b.h[1] = *(const h8*)(&Kc[off + 16]);
                sc[nb] = __builtin_amdgcn_wmma_f32_16x16x32_f16(
                    false, a.v, false, b.v, (short)0, sc[nb], false, false);
            }
        }

        // ---- retire K staging registers into the shadow buffer -----------
        if (haveNext) {
            #pragma unroll
            for (int i = 0; i < 8; ++i) {
                const int fi  = tid + i * NTHR;
                const int row = fi >> 5;
                const int col = (fi & 31) << 2;
                h4 kh;
                kh[0] = (_Float16)kreg[i].x; kh[1] = (_Float16)kreg[i].y;
                kh[2] = (_Float16)kreg[i].z; kh[3] = (_Float16)kreg[i].w;
                *(h4*)(&Kn[row * KPAD + col]) = kh;
            }
        }

        // ---- phase 2: issue next V-tile loads (32 staging VGPRs) ---------
        float4 vreg[8];
        if (haveNext) {
            const float4* vg4 = (const float4*)(vgb + (size_t)(s0 + BLOCK_N) * DHEAD);
            #pragma unroll
            for (int i = 0; i < 8; ++i) vreg[i] = vg4[tid + i * NTHR];
        }

        // ---- mask + scale + online softmax -------------------------------
        #pragma unroll
        for (int v = 0; v < 8; ++v) {
            const int rm = v + 8 * hi;
            const float* mp = Mask + (size_t)(qrow0 + rm) * SK + s0 + ln;
            float t0 = (sc[0][v] + mp[0])  * scale;
            float t1 = (sc[1][v] + mp[16]) * scale;
            float t2 = (sc[2][v] + mp[32]) * scale;
            float t3 = (sc[3][v] + mp[48]) * scale;
            float pm = fmaxf(fmaxf(t0, t1), fmaxf(t2, t3));
            #pragma unroll
            for (int off = 1; off < 16; off <<= 1)
                pm = fmaxf(pm, __shfl_xor(pm, off, 32));
            const float mo = mrow[v];
            const float mn = fmaxf(mo, pm);
            float p0 = exp2f((t0 - mn) * L2E);
            float p1 = exp2f((t1 - mn) * L2E);
            float p2 = exp2f((t2 - mn) * L2E);
            float p3 = exp2f((t3 - mn) * L2E);
            float rs = (p0 + p1) + (p2 + p3);
            #pragma unroll
            for (int off = 1; off < 16; off <<= 1)
                rs += __shfl_xor(rs, off, 32);
            const float alpha = exp2f((mo - mn) * L2E);
            mrow[v] = mn;
            lrow[v] = lrow[v] * alpha + rs;
            #pragma unroll
            for (int o = 0; o < 8; ++o) acc[o][v] *= alpha;
            _Float16* pp = &Psh[(wave * 16 + rm) * PPAD + ln];
            pp[0]  = (_Float16)p0;
            pp[16] = (_Float16)p1;
            pp[32] = (_Float16)p2;
            pp[48] = (_Float16)p3;
        }

        // ---- O += P * V  (A from wave-private P buffer, B from V^T) ------
        #pragma unroll
        for (int kc2 = 0; kc2 < 2; ++kc2) {
            fragA ap;
            const int pb = (wave * 16 + ln) * PPAD + kc2 * 32 + hi * 8;
            ap.h[0] = *(const h8*)(&Psh[pb]);
            ap.h[1] = *(const h8*)(&Psh[pb + 16]);
            #pragma unroll
            for (int nb2 = 0; nb2 < 8; ++nb2) {
                fragA bv;
                const int off = (nb2 * 16 + ln) * VPAD + kc2 * 32 + hi * 8;
                bv.h[0] = *(const h8*)(&Vc[off]);
                bv.h[1] = *(const h8*)(&Vc[off + 16]);
                acc[nb2] = __builtin_amdgcn_wmma_f32_16x16x32_f16(
                    false, ap.v, false, bv.v, (short)0, acc[nb2], false, false);
            }
        }

        // ---- retire V staging registers into the shadow buffer -----------
        if (haveNext) {
            #pragma unroll
            for (int i = 0; i < 8; ++i) {
                const int fi  = tid + i * NTHR;
                const int row = fi >> 5;
                const int col = (fi & 31) << 2;
                Vn[(col + 0) * VPAD + row] = (_Float16)vreg[i].x;
                Vn[(col + 1) * VPAD + row] = (_Float16)vreg[i].y;
                Vn[(col + 2) * VPAD + row] = (_Float16)vreg[i].z;
                Vn[(col + 3) * VPAD + row] = (_Float16)vreg[i].w;
            }
        }
        __syncthreads();   // next iter reads the freshly written buffers
    }

    // ---- finalize: divide by softmax denominator, write f32 --------------
    float* orow = O + ((size_t)bh * LQ + qrow0) * DHEAD;
    #pragma unroll
    for (int v = 0; v < 8; ++v) {
        const float inv = 1.0f / lrow[v];
        const int rm = v + 8 * hi;
        #pragma unroll
        for (int nb2 = 0; nb2 < 8; ++nb2)
            orow[rm * DHEAD + nb2 * 16 + ln] = acc[nb2][v] * inv;
    }
}

extern "C" void kernel_launch(void* const* d_in, const int* in_sizes, int n_in,
                              void* d_out, int out_size, void* d_ws, size_t ws_size,
                              hipStream_t stream) {
    (void)in_sizes; (void)n_in; (void)out_size; (void)d_ws; (void)ws_size;
    const float* q    = (const float*)d_in[0];
    const float* k    = (const float*)d_in[1];
    const float* v    = (const float*)d_in[2];
    const float* mask = (const float*)d_in[3];
    float* out = (float*)d_out;
    dim3 grid(LQ / BLOCK_M, NBH);
    fa_fwd_f16wmma<<<grid, NTHR, 0, stream>>>(q, k, v, mask, out);
}
